// Drug_GAT_29583734735544
// MI455X (gfx1250) — matrix operand order repaired and checked by reading, not compiled, not run on hardware
//
#include <hip/hip_runtime.h>
#include <hip/hip_bf16.h>

// ---------------- problem constants ----------------
#define NNODES 50000
#define NEDGES 400000
#define NGRAPH 256
#define IN_DIM 79
#define HEADS 8
#define OUT_DIM 128
#define HC 1024            // HEADS*OUT_DIM
#define KPAD 96            // IN_DIM padded to 3*32
#define MT 3125            // NNODES/16 (exact)
#define NTL 64             // HC/16
#define NTG 16             // groups of 4 N-tiles
#define KT 3               // KPAD/32
#define EP (NEDGES + NNODES)   // edges + self loops = 450000
#define NEG_SLOPE 0.2f

typedef __attribute__((ext_vector_type(16))) _Float16 v16h;
typedef __attribute__((ext_vector_type(8)))  float    v8f;
typedef __attribute__((ext_vector_type(4)))  unsigned u32x4;
typedef __attribute__((ext_vector_type(4)))  int      i32x4;
typedef __attribute__((ext_vector_type(8)))  int      i32x8;

// ---- WMMA fragment K maps (CDNA5 ISA §7.12.2) ----
// A 16x32 f16: lane(0-15)=row M, elems 0..7 -> K = kb+e, elems 8..15 -> K = 16+kb+(e-8), kb = 8*(lane>=16)
__device__ __forceinline__ int kmapA(int lane, int e) {
    return (e < 8 ? e : 8 + e) + ((lane >> 4) << 3);
}
// B 32x16 f16: lane(0-15)=col N, same K map as A: elems 0..7 -> K = kb+e, elems 8..15 -> K = 16+kb+(e-8), kb = 8*(lane>=16)
__device__ __forceinline__ int kmapB(int lane, int e) {
    return (e < 8 ? e : 8 + e) + ((lane >> 4) << 3);
}

// order-preserving float<->uint encoding for atomic max
__device__ __forceinline__ unsigned fenc(float f) {
    unsigned u = __float_as_uint(f);
    return (u & 0x80000000u) ? ~u : (u | 0x80000000u);
}
__device__ __forceinline__ float fdec(unsigned u) {
    u = (u & 0x80000000u) ? (u & 0x7FFFFFFFu) : ~u;
    return __uint_as_float(u);
}

// ---------------- kernels ----------------

__global__ void k_zero(unsigned* __restrict__ p, size_t nwords) {
    size_t i = (size_t)blockIdx.x * blockDim.x + threadIdx.x;
    size_t stride = (size_t)gridDim.x * blockDim.x;
    for (; i < nwords; i += stride) p[i] = 0u;
}

// pack x [N, 79] f32 -> A fragments [MT][KT][32 lanes][16 halves]
__global__ void k_pack_a(const float* __restrict__ x, _Float16* __restrict__ A) {
    int t = blockIdx.x * blockDim.x + threadIdx.x;   // one thread per (mt, ks, lane)
    if (t >= MT * KT * 32) return;
    int lane = t & 31;
    int ks = (t >> 5) % KT;
    int mt = t / (KT * 32);
    int m = mt * 16 + (lane & 15);
    v16h v;
#pragma unroll
    for (int e = 0; e < 16; ++e) {
        int k = kmapA(lane, e) + 32 * ks;
        float f = (k < IN_DIM) ? x[(size_t)m * IN_DIM + k] : 0.f;
        v[e] = (_Float16)f;
    }
    *(v16h*)(A + (size_t)t * 16) = v;
}

// pack lin_w [79, 1024] f32 -> B fragments [NTL][KT][32 lanes][16 halves]
// (4 consecutive nt = one contiguous 12 KB chunk = one TDM tile)
__global__ void k_pack_b(const float* __restrict__ w, _Float16* __restrict__ B) {
    int t = blockIdx.x * blockDim.x + threadIdx.x;   // one thread per (nt, ks, lane)
    if (t >= NTL * KT * 32) return;
    int lane = t & 31;
    int ks = (t >> 5) % KT;
    int nt = t / (KT * 32);
    int n = nt * 16 + (lane & 15);
    v16h v;
#pragma unroll
    for (int e = 0; e < 16; ++e) {
        int k = kmapB(lane, e) + 32 * ks;
        float f = (k < IN_DIM) ? w[(size_t)k * HC + n] : 0.f;
        v[e] = (_Float16)f;
    }
    *(v16h*)(B + (size_t)t * 16) = v;
}

// H = x @ lin_w. Block = 8 waves; block covers 8 M-tiles x 4 N-tiles.
// B fragments for the 4 N-tiles (12 KB) are staged in LDS once per block via the
// Tensor Data Mover (tensor_load_to_lds + s_wait_tensorcnt), then each wave does
// 12 WMMAs reusing one A fragment per k-step across 4 accumulators.
__global__ void k_gemm(const _Float16* __restrict__ A, const _Float16* __restrict__ B,
                       float* __restrict__ H) {
    __shared__ __attribute__((aligned(32))) _Float16 ldsB[4 * KT * 32 * 16]; // 12 KB
    int wave = threadIdx.x >> 5;
    int lane = threadIdx.x & 31;
    int ntg = blockIdx.y;                 // which group of 4 N-tiles
    int mt = blockIdx.x * 8 + wave;       // this wave's M tile

#if defined(__gfx1250__) && __has_builtin(__builtin_amdgcn_tensor_load_to_lds)
    if (threadIdx.x < 32) {               // wave 0 issues the TDM DMA (EXEC ignored by TDM)
        unsigned lds_addr = (unsigned)(uintptr_t)(void*)ldsB;   // addr[31:0] = LDS byte offset
        unsigned long long ga = (unsigned long long)(uintptr_t)(B + (size_t)ntg * 6144);
        // D# group 0: count=1 | lds_addr | global_addr[56:0] | type=2 (bits 127:126)
        u32x4 g0 = {};
        g0[0] = 1u;
        g0[1] = lds_addr;
        g0[2] = (unsigned)ga;
        g0[3] = ((unsigned)(ga >> 32) & 0x01FFFFFFu) | 0x80000000u;
        // D# group 1: 1-D copy, data_size=8B, 1536 elements (= 12 KB), stride = extent
        i32x8 g1 = {};
        g1[0] = (int)(3u << 16);                 // data_size = 8 bytes
        g1[1] = (int)((1536u & 0xFFFFu) << 16);  // tensor_dim0[15:0]
        g1[2] = (int)(1u << 16);                 // tensor_dim0[31:16]=0, tensor_dim1=1
        g1[3] = (int)(1536u << 16);              // tile_dim0 = 1536
        g1[4] = 1;                               // tile_dim1 = 1
        g1[5] = 1536;                            // tensor_dim0_stride
        i32x4 g2 = {}, g3 = {};
        i32x8 g4 = {};
        __builtin_amdgcn_tensor_load_to_lds(g0, g1, g2, g3, g4, 0);
        __builtin_amdgcn_s_wait_tensorcnt(0);
    }
#else
    {   // cooperative fallback: 256 threads copy 1536 x 8B
        const unsigned long long* src = (const unsigned long long*)(B + (size_t)ntg * 6144);
        unsigned long long* dst = (unsigned long long*)ldsB;
        for (int i = threadIdx.x; i < 1536; i += 256) dst[i] = src[i];
    }
#endif
    __syncthreads();

    if (mt < MT) {                        // wave-uniform guard: EXEC all-ones inside
        v8f acc0 = {}, acc1 = {}, acc2 = {}, acc3 = {};
        const _Float16* ap = A + (size_t)mt * KT * 32 * 16;
#pragma unroll
        for (int ks = 0; ks < KT; ++ks) {
            v16h a = *(const v16h*)(ap + (ks * 32 + lane) * 16);
            v16h b0 = *(const v16h*)(ldsB + ((0 * KT + ks) * 32 + lane) * 16);
            v16h b1 = *(const v16h*)(ldsB + ((1 * KT + ks) * 32 + lane) * 16);
            v16h b2 = *(const v16h*)(ldsB + ((2 * KT + ks) * 32 + lane) * 16);
            v16h b3 = *(const v16h*)(ldsB + ((3 * KT + ks) * 32 + lane) * 16);
            acc0 = __builtin_amdgcn_wmma_f32_16x16x32_f16(false, a, false, b0, (short)0, acc0, false, false);
            acc1 = __builtin_amdgcn_wmma_f32_16x16x32_f16(false, a, false, b1, (short)0, acc1, false, false);
            acc2 = __builtin_amdgcn_wmma_f32_16x16x32_f16(false, a, false, b2, (short)0, acc2, false, false);
            acc3 = __builtin_amdgcn_wmma_f32_16x16x32_f16(false, a, false, b3, (short)0, acc3, false, false);
        }
        int nlo = lane & 15;
        int mbase = mt * 16 + ((lane >> 4) << 3);
        float* hb = H + (size_t)mbase * HC + (ntg * 4) * 16 + nlo;
#pragma unroll
        for (int r = 0; r < 8; ++r) {
            hb[(size_t)r * HC + 0]  = acc0[r];
            hb[(size_t)r * HC + 16] = acc1[r];
            hb[(size_t)r * HC + 32] = acc2[r];
            hb[(size_t)r * HC + 48] = acc3[r];
        }
    }
}

// a_src[n,h] = dot(H[n,h,:], att_src[h,:]);  block per node, wave per head
__global__ void k_att(const float* __restrict__ H, const float* __restrict__ att_s,
                      const float* __restrict__ att_d, float* __restrict__ aS,
                      float* __restrict__ aD) {
    int n = blockIdx.x;
    int wave = threadIdx.x >> 5, lane = threadIdx.x & 31;
    float4 hv = *(const float4*)(H + (size_t)n * HC + wave * OUT_DIM + lane * 4);
    float4 as = *(const float4*)(att_s + wave * OUT_DIM + lane * 4);
    float4 ad = *(const float4*)(att_d + wave * OUT_DIM + lane * 4);
    float s = hv.x * as.x + hv.y * as.y + hv.z * as.z + hv.w * as.w;
    float d = hv.x * ad.x + hv.y * ad.y + hv.z * ad.z + hv.w * ad.w;
#pragma unroll
    for (int off = 16; off; off >>= 1) {
        s += __shfl_xor(s, off, 32);
        d += __shfl_xor(d, off, 32);
    }
    if (lane == 0) {
        aS[n * HEADS + wave] = s;
        aD[n * HEADS + wave] = d;
    }
}

__device__ __forceinline__ void edge_ids(int e, const int* __restrict__ ei, int& src, int& dst) {
    if (e < NEDGES) { src = ei[e]; dst = ei[NEDGES + e]; }
    else            { src = dst = e - NEDGES; }
}

// pass 1: segment max of leaky_relu logits via atomicMax on encoded uint
__global__ void k_edge_max(const int* __restrict__ ei, const float* __restrict__ aS,
                           const float* __restrict__ aD, unsigned* __restrict__ mx) {
    int t = blockIdx.x * blockDim.x + threadIdx.x;
    if (t >= EP * HEADS) return;
    int e = t >> 3, h = t & 7;
    int src, dst; edge_ids(e, ei, src, dst);
    float lo = aS[src * HEADS + h] + aD[dst * HEADS + h];
    lo = lo > 0.f ? lo : NEG_SLOPE * lo;
    atomicMax(mx + dst * HEADS + h, fenc(lo));
}

// pass 2: exp(logit - max), store per-edge, segment-sum via atomicAdd
__global__ void k_edge_exp(const int* __restrict__ ei, const float* __restrict__ aS,
                           const float* __restrict__ aD, const unsigned* __restrict__ mx,
                           float* __restrict__ ee, float* __restrict__ sm) {
    int t = blockIdx.x * blockDim.x + threadIdx.x;
    if (t >= EP * HEADS) return;
    int e = t >> 3, h = t & 7;
    int src, dst; edge_ids(e, ei, src, dst);
    float lo = aS[src * HEADS + h] + aD[dst * HEADS + h];
    lo = lo > 0.f ? lo : NEG_SLOPE * lo;
    float ev = expf(lo - fdec(mx[dst * HEADS + h]));
    ee[(size_t)e * HEADS + h] = ev;
    atomicAdd(sm + dst * HEADS + h, ev);
}

// pass 3: out[dst] += alpha * H[src]; block per edge, channel-major coalesced
__global__ void k_msg(const int* __restrict__ ei, const float* __restrict__ H,
                      const float* __restrict__ ee, const float* __restrict__ sm,
                      float* __restrict__ outf) {
    int e = blockIdx.x;
    int tid = threadIdx.x;
    int src, dst; edge_ids(e, ei, src, dst);
    __shared__ float salpha[HEADS];
    if (tid < HEADS)
        salpha[tid] = ee[(size_t)e * HEADS + tid] / (sm[dst * HEADS + tid] + 1e-16f);
    __syncthreads();
    const float* hs = H + (size_t)src * HC;
    float* od = outf + (size_t)dst * HC;
#pragma unroll
    for (int i = 0; i < 4; ++i) {
        int hc = tid + 256 * i;
        atomicAdd(od + hc, hs[hc] * salpha[hc >> 7]);
    }
}

// relu(out + bias), accumulate into per-graph pool
__global__ void k_pool(const float* __restrict__ outf, const float* __restrict__ bias,
                       const int* __restrict__ batch, float* __restrict__ pooled,
                       float* __restrict__ cnt) {
    int n = blockIdx.x, tid = threadIdx.x;
    int b = batch[n];
#pragma unroll
    for (int i = 0; i < 4; ++i) {
        int hc = tid + 256 * i;
        float v = outf[(size_t)n * HC + hc] + bias[hc];
        v = v > 0.f ? v : 0.f;
        atomicAdd(pooled + (size_t)b * HC + hc, v);
    }
    if (tid == 0) atomicAdd(cnt + b, 1.0f);
}

// (pooled/cnt) @ fc_w + fc_b; block per graph
__global__ void k_final(const float* __restrict__ pooled, const float* __restrict__ cnt,
                        const float* __restrict__ fc_w, const float* __restrict__ fc_b,
                        float* __restrict__ out) {
    int g = blockIdx.x, tid = threadIdx.x;
    float p = 0.f;
#pragma unroll
    for (int i = 0; i < 4; ++i) {
        int hc = tid + 256 * i;
        p += pooled[(size_t)g * HC + hc] * fc_w[hc];
    }
#pragma unroll
    for (int off = 16; off; off >>= 1) p += __shfl_xor(p, off, 32);
    __shared__ float ws[8];
    if ((tid & 31) == 0) ws[tid >> 5] = p;
    __syncthreads();
    if (tid == 0) {
        float s = 0.f;
#pragma unroll
        for (int w = 0; w < 8; ++w) s += ws[w];
        float c = cnt[g]; c = c > 1.f ? c : 1.f;
        out[g] = s / c + fc_b[0];
    }
}

// ---------------- launcher ----------------
extern "C" void kernel_launch(void* const* d_in, const int* in_sizes, int n_in,
                              void* d_out, int out_size, void* d_ws, size_t ws_size,
                              hipStream_t stream) {
    const float* x      = (const float*)d_in[0];
    const int*   ei     = (const int*)d_in[1];
    const int*   batch  = (const int*)d_in[2];
    const float* lin_w  = (const float*)d_in[3];
    const float* att_s  = (const float*)d_in[4];
    const float* att_d  = (const float*)d_in[5];
    const float* bias   = (const float*)d_in[6];
    const float* fc_w   = (const float*)d_in[7];
    const float* fc_b   = (const float*)d_in[8];
    float* out = (float*)d_out;

    char* base = (char*)d_ws;
    size_t off = 0;
    auto take = [&](size_t bytes) -> char* {
        char* p = base + off;
        off = (off + bytes + 255) & ~(size_t)255;
        return p;
    };
    _Float16* Afrag = (_Float16*)take((size_t)MT * KT * 32 * 16 * 2);   // 9.6 MB
    _Float16* Bfrag = (_Float16*)take((size_t)NTL * KT * 32 * 16 * 2);  // 0.2 MB
    float*    H     = (float*)take((size_t)NNODES * HC * 4);            // 204.8 MB
    float*    aS    = (float*)take((size_t)NNODES * HEADS * 4);
    float*    aD    = (float*)take((size_t)NNODES * HEADS * 4);
    float*    ee    = (float*)take((size_t)EP * HEADS * 4);             // 14.4 MB
    // contiguous zero-init region:
    size_t zbeg = off;
    unsigned* mx    = (unsigned*)take((size_t)NNODES * HEADS * 4);
    float*    sm    = (float*)take((size_t)NNODES * HEADS * 4);
    float*    outf  = (float*)take((size_t)NNODES * HC * 4);            // 204.8 MB
    float*    pooled= (float*)take((size_t)NGRAPH * HC * 4);
    float*    cnt   = (float*)take((size_t)NGRAPH * 4);
    size_t zwords = (off - zbeg) / 4;

    (void)in_sizes; (void)n_in; (void)out_size; (void)ws_size;

    // 0) zero accumulators / maxima
    k_zero<<<2048, 256, 0, stream>>>((unsigned*)(base + zbeg), zwords);

    // 1) pack WMMA fragments
    k_pack_a<<<(MT * KT * 32 + 255) / 256, 256, 0, stream>>>(x, Afrag);
    k_pack_b<<<(NTL * KT * 32 + 255) / 256, 256, 0, stream>>>(lin_w, Bfrag);

    // 2) H = x @ lin_w : blocks of 8 M-tiles x (4 N-tiles staged via TDM into LDS)
    {
        dim3 g((MT + 7) / 8, NTG);
        k_gemm<<<g, 256, 0, stream>>>(Afrag, Bfrag, H);
    }

    // 3) attention coefficients
    k_att<<<NNODES, 256, 0, stream>>>(H, att_s, att_d, aS, aD);

    // 4) segment softmax (3 passes) + message scatter
    int ethreads = EP * HEADS;
    k_edge_max<<<(ethreads + 255) / 256, 256, 0, stream>>>(ei, aS, aD, mx);
    k_edge_exp<<<(ethreads + 255) / 256, 256, 0, stream>>>(ei, aS, aD, mx, ee, sm);
    k_msg<<<EP, 256, 0, stream>>>(ei, H, ee, sm, outf);

    // 5) bias + relu + per-graph mean pool, then FC
    k_pool<<<NNODES, 256, 0, stream>>>(outf, bias, batch, pooled, cnt);
    k_final<<<NGRAPH, 256, 0, stream>>>(pooled, cnt, fc_w, fc_b, out);
}
